// Self_Attention_27659589386606
// MI455X (gfx1250) — matrix operand order repaired and checked by reading, not compiled, not run on hardware
//
#include <hip/hip_runtime.h>

typedef __attribute__((ext_vector_type(16))) _Float16 v16h;
typedef __attribute__((ext_vector_type(8)))  _Float16 v8h;
typedef __attribute__((ext_vector_type(8)))  float    v8f;

union v16h_pair { v16h v; v8h h[2]; };

#define B_ 4
#define T_ 2048
#define D_ 1024
#define H_ 64

// ---- CDNA5 async global->LDS copy (ASYNCcnt) ------------------------------
__device__ __forceinline__ void async_ld_b128(unsigned lds_off, unsigned voff,
                                              const void* sbase) {
  asm volatile("global_load_async_to_lds_b128 %0, %1, %2 offset:0"
               :: "v"(lds_off), "v"(voff), "s"(sbase)
               : "memory");
}
__device__ __forceinline__ void wait_async0() {
  asm volatile("s_wait_asynccnt 0" ::: "memory");
}
__device__ __forceinline__ unsigned lds_addr(const void* p) {
  return (unsigned)(uintptr_t)p;          // LDS aperture: offset lives in addr[31:0]
}

// reduce across the 16-lane half of a wave32 (xor masks 1,2,4,8 stay within halves)
__device__ __forceinline__ float half16_max(float v) {
  v = fmaxf(v, __shfl_xor(v, 1));
  v = fmaxf(v, __shfl_xor(v, 2));
  v = fmaxf(v, __shfl_xor(v, 4));
  v = fmaxf(v, __shfl_xor(v, 8));
  return v;
}
__device__ __forceinline__ float half16_sum(float v) {
  v += __shfl_xor(v, 1);
  v += __shfl_xor(v, 2);
  v += __shfl_xor(v, 4);
  v += __shfl_xor(v, 8);
  return v;
}

// ---------------------------------------------------------------------------
// Kernel 0: transpose-convert the three weight matrices: W[d][h] f32 ->
// WT[h][d] f16, so projection B-fragments become contiguous per-lane loads.
// ---------------------------------------------------------------------------
__global__ __launch_bounds__(256)
void wconv_kernel(const float* __restrict__ Wq, const float* __restrict__ Wk,
                  const float* __restrict__ Wv, _Float16* __restrict__ WqT,
                  _Float16* __restrict__ WkT, _Float16* __restrict__ WvT) {
  const int idx = blockIdx.x * 256 + threadIdx.x;   // 0 .. D*H-1
  const int h = idx / D_;
  const int d = idx % D_;
  const size_t src = (size_t)d * H_ + h;
  WqT[idx] = (_Float16)Wq[src];
  WkT[idx] = (_Float16)Wk[src];
  WvT[idx] = (_Float16)Wv[src];
}

// ---------------------------------------------------------------------------
// Kernel 1: QKV projection.  8 waves/block share async-staged W tiles in LDS;
// each wave produces a 16x64 tile of Q, K, V via v_wmma_f32_16x16x32_f16.
// Q,K stored [token][h]; V stored transposed [h][token] (f16).
// ---------------------------------------------------------------------------
__global__ __launch_bounds__(256)
void qkv_proj_kernel(const float* __restrict__ x,
                     const _Float16* __restrict__ WqT,
                     const _Float16* __restrict__ WkT,
                     const _Float16* __restrict__ WvT,
                     _Float16* __restrict__ Qh,
                     _Float16* __restrict__ Kh,
                     _Float16* __restrict__ Vt) {
  // [buf][matrix][h-col 0..63][k-local 0..31] : 2 x 12KB
  __shared__ __align__(32) _Float16 Wlds[2][3][64][32];

  const int tid  = threadIdx.x;
  const int lane = tid & 31;
  const int wave = tid >> 5;
  const int tile = blockIdx.x * 8 + wave;          // 0..511
  const int tilesPerB = T_ / 16;                   // 128
  const int b    = tile / tilesPerB;
  const int row0 = (tile % tilesPerB) * 16;
  const int m    = lane & 15;                      // A row / B col index
  const int half = lane >> 4;

  // Per-thread staging slice: row = tid/4 (0..63), chunk = tid%4 (8 f16 = 16B)
  const int srow = tid >> 2;
  const int scol = (tid & 3) * 8;
  const void* wsrc[3] = { WqT, WkT, WvT };

  auto stage_w = [&](int kt, int buf) {
    const unsigned gof = (unsigned)((srow * D_ + kt + scol) * sizeof(_Float16));
#pragma unroll
    for (int w = 0; w < 3; ++w)
      async_ld_b128(lds_addr(&Wlds[buf][w][srow][scol]), gof, wsrc[w]);
  };

  v8f accQ[4], accK[4], accV[4];
#pragma unroll
  for (int nt = 0; nt < 4; ++nt) { accQ[nt] = {}; accK[nt] = {}; accV[nt] = {}; }

  const float* xrow = x + ((size_t)(b * T_ + row0 + m)) * D_;

  stage_w(0, 0);
  wait_async0();
  __syncthreads();
  int buf = 0;

  for (int kt = 0; kt < D_; kt += 32) {
    if (kt + 32 < D_) stage_w(kt + 32, buf ^ 1);   // overlap next tile

    // A fragment: two contiguous 8-float runs -> b128 loads + packed cvt
    const v8f x0 = *(const v8f*)(xrow + kt + half * 16);
    const v8f x1 = *(const v8f*)(xrow + kt + half * 16 + 8);
    v16h a;
#pragma unroll
    for (int i = 0; i < 8; ++i) {
      a[i]     = (_Float16)x0[i];
      a[8 + i] = (_Float16)x1[i];
    }
#pragma unroll
    for (int nt = 0; nt < 4; ++nt) {
      const int col = nt * 16 + m;
      const v16h bq = *(const v16h*)(&Wlds[buf][0][col][half * 16]);  // ds b128 x2
      const v16h bk = *(const v16h*)(&Wlds[buf][1][col][half * 16]);
      const v16h bv = *(const v16h*)(&Wlds[buf][2][col][half * 16]);
      accQ[nt] = __builtin_amdgcn_wmma_f32_16x16x32_f16(false, a, false, bq,
                                                        (short)0, accQ[nt], false, false);
      accK[nt] = __builtin_amdgcn_wmma_f32_16x16x32_f16(false, a, false, bk,
                                                        (short)0, accK[nt], false, false);
      accV[nt] = __builtin_amdgcn_wmma_f32_16x16x32_f16(false, a, false, bv,
                                                        (short)0, accV[nt], false, false);
    }

    wait_async0();
    __syncthreads();
    buf ^= 1;
  }

  // C/D layout: VGPR r -> row M = r + 8*half, col N = lane&15
#pragma unroll
  for (int nt = 0; nt < 4; ++nt) {
#pragma unroll
    for (int r = 0; r < 8; ++r) {
      const int M = r + 8 * half;
      const size_t idx = ((size_t)(b * T_ + row0 + M)) * H_ + nt * 16 + m;
      Qh[idx] = (_Float16)accQ[nt][r];
      Kh[idx] = (_Float16)accK[nt][r];
    }
    // V transposed: lane's 8 rows are consecutive keys -> one b128 store
    v8h pv;
#pragma unroll
    for (int r = 0; r < 8; ++r) pv[r] = (_Float16)accV[nt][r];
    *(v8h*)(Vt + ((size_t)b * H_ + nt * 16 + m) * T_ + row0 + 8 * half) = pv;
  }
}

// ---------------------------------------------------------------------------
// Kernel 2: flash attention.  A block's 8 waves share one batch and the same
// key sweep, so K/V tiles are async-staged to LDS once per block (8x traffic
// cut), double-buffered against the WMMA work.
// ---------------------------------------------------------------------------
__global__ __launch_bounds__(256)
void flash_attn_kernel(const _Float16* __restrict__ Qh,
                       const _Float16* __restrict__ Kh,
                       const _Float16* __restrict__ Vt,
                       float* __restrict__ out) {
  __shared__ __align__(32) _Float16 Klds[2][32][64];   // 2 x 4KB
  __shared__ __align__(32) _Float16 Vlds[2][64][32];   // 2 x 4KB
  __shared__ __align__(32) _Float16 Plds[8][16][32];   // per-wave P staging, 8KB

  const int tid  = threadIdx.x;
  const int lane = tid & 31;
  const int wave = tid >> 5;
  const int tile = blockIdx.x * 8 + wave;          // 0..511
  const int tilesPerB = T_ / 16;
  const int b    = tile / tilesPerB;
  const int row0 = (tile % tilesPerB) * 16;
  const int m    = lane & 15;
  const int half = lane >> 4;

  const _Float16* kb  = Kh + (size_t)b * T_ * H_;
  const _Float16* vtb = Vt + (size_t)b * H_ * T_;

  // Staging slices: K tile 32x64 (thread: row tid/8, chunk tid%8);
  //                 V tile 64x32 (thread: row tid/4, chunk tid%4)
  const int krow = tid >> 3, kcol = (tid & 7) * 8;
  const int vrow = tid >> 2, vcol = (tid & 3) * 8;

  auto stage_kv = [&](int kt, int buf) {
    async_ld_b128(lds_addr(&Klds[buf][krow][kcol]),
                  (unsigned)(((kt + krow) * H_ + kcol) * sizeof(_Float16)), kb);
    async_ld_b128(lds_addr(&Vlds[buf][vrow][vcol]),
                  (unsigned)((vrow * T_ + kt + vcol) * sizeof(_Float16)), vtb);
  };

  // Resident Q A-fragments (h 0..31 and 32..63): four b128 loads
  v16h aq0, aq1;
  {
    const _Float16* qrow = Qh + ((size_t)(b * T_ + row0 + m)) * H_;
    v16h_pair t0, t1;
    t0.h[0] = *(const v8h*)(qrow + half * 16);
    t0.h[1] = *(const v8h*)(qrow + half * 16 + 8);
    t1.h[0] = *(const v8h*)(qrow + 32 + half * 16);
    t1.h[1] = *(const v8h*)(qrow + 32 + half * 16 + 8);
    aq0 = t0.v;
    aq1 = t1.v;
  }

  const float scale = 0.125f;                      // 1/sqrt(64)
  v8f O[4];
#pragma unroll
  for (int nt = 0; nt < 4; ++nt) O[nt] = {};
  float mrow[8], lrow[8];
#pragma unroll
  for (int r = 0; r < 8; ++r) { mrow[r] = -3.0e38f; lrow[r] = 0.0f; }

  stage_kv(0, 0);
  wait_async0();
  __syncthreads();
  int buf = 0;

  for (int kt = 0; kt < T_; kt += 32) {
    if (kt + 32 < T_) stage_kv(kt + 32, buf ^ 1);  // overlap next tile

    // B fragments of K^T from LDS: per-lane contiguous 32B runs (ds b128 x2)
    const _Float16* k0 = &Klds[buf][m][half * 16];
    const _Float16* k1 = &Klds[buf][16 + m][half * 16];
    const v16h bk0lo = *(const v16h*)(k0);
    const v16h bk0hi = *(const v16h*)(k0 + 32);
    const v16h bk1lo = *(const v16h*)(k1);
    const v16h bk1hi = *(const v16h*)(k1 + 32);

    v8f S0 = {}, S1 = {};
    S0 = __builtin_amdgcn_wmma_f32_16x16x32_f16(false, aq0, false, bk0lo, (short)0, S0, false, false);
    S0 = __builtin_amdgcn_wmma_f32_16x16x32_f16(false, aq1, false, bk0hi, (short)0, S0, false, false);
    S1 = __builtin_amdgcn_wmma_f32_16x16x32_f16(false, aq0, false, bk1lo, (short)0, S1, false, false);
    S1 = __builtin_amdgcn_wmma_f32_16x16x32_f16(false, aq1, false, bk1hi, (short)0, S1, false, false);

    // Online softmax update (row stats replicated across each 16-lane half)
    float corr[8];
#pragma unroll
    for (int r = 0; r < 8; ++r) {
      const float s0 = S0[r] * scale;
      const float s1 = S1[r] * scale;
      const float rm   = half16_max(fmaxf(s0, s1));
      const float mnew = fmaxf(mrow[r], rm);
      const float c    = __expf(mrow[r] - mnew);
      const float p0   = __expf(s0 - mnew);
      const float p1   = __expf(s1 - mnew);
      S0[r] = p0;
      S1[r] = p1;
      lrow[r] = lrow[r] * c + half16_sum(p0 + p1);
      mrow[r] = mnew;
      corr[r] = c;
    }
#pragma unroll
    for (int nt = 0; nt < 4; ++nt)
#pragma unroll
      for (int r = 0; r < 8; ++r) O[nt][r] *= corr[r];

    // Transpose P (C-layout -> A-layout) through this wave's LDS tile
#pragma unroll
    for (int r = 0; r < 8; ++r) {
      Plds[wave][8 * half + r][m]      = (_Float16)S0[r];
      Plds[wave][8 * half + r][m + 16] = (_Float16)S1[r];
    }
    asm volatile("s_wait_dscnt 0" ::: "memory");   // order same-wave DS store->load

    v16h_pair tp;                                   // two ds_load_b128
    tp.h[0] = *(const v8h*)(&Plds[wave][m][half * 16]);
    tp.h[1] = *(const v8h*)(&Plds[wave][m][half * 16 + 8]);
    const v16h ap = tp.v;

    // O += P * V from LDS (ds b128 x2 per slab)
#pragma unroll
    for (int nt = 0; nt < 4; ++nt) {
      const v16h bv = *(const v16h*)(&Vlds[buf][nt * 16 + m][half * 16]);
      O[nt] = __builtin_amdgcn_wmma_f32_16x16x32_f16(false, ap, false, bv,
                                                     (short)0, O[nt], false, false);
    }

    wait_async0();
    __syncthreads();
    buf ^= 1;
  }

  // Epilogue: normalize and store f32
#pragma unroll
  for (int nt = 0; nt < 4; ++nt) {
#pragma unroll
    for (int r = 0; r < 8; ++r) {
      const int M = r + 8 * half;
      out[((size_t)(b * T_ + row0 + M)) * H_ + nt * 16 + m] = O[nt][r] / lrow[r];
    }
  }
}

// ---------------------------------------------------------------------------
extern "C" void kernel_launch(void* const* d_in, const int* in_sizes, int n_in,
                              void* d_out, int out_size, void* d_ws, size_t ws_size,
                              hipStream_t stream) {
  const float* x  = (const float*)d_in[0];
  const float* Wq = (const float*)d_in[1];
  const float* Wk = (const float*)d_in[2];
  const float* Wv = (const float*)d_in[3];

  const size_t nQKV = (size_t)B_ * T_ * H_;        // 524288 f16 each
  const size_t nW   = (size_t)D_ * H_;             // 65536 f16 each
  _Float16* Qh  = (_Float16*)d_ws;
  _Float16* Kh  = Qh + nQKV;
  _Float16* Vt  = Kh + nQKV;                       // transposed [b][h][key]
  _Float16* WqT = Vt + nQKV;
  _Float16* WkT = WqT + nW;
  _Float16* WvT = WkT + nW;                        // total ~3.4 MB scratch
  float* out = (float*)d_out;

  wconv_kernel<<<dim3((D_ * H_) / 256), dim3(256), 0, stream>>>(Wq, Wk, Wv, WqT, WkT, WvT);
  qkv_proj_kernel<<<dim3(64), dim3(256), 0, stream>>>(x, WqT, WkT, WvT, Qh, Kh, Vt);
  flash_attn_kernel<<<dim3(64), dim3(256), 0, stream>>>(Qh, Kh, Vt, out);
}
